// NNConv_26216480375300
// MI455X (gfx1250) — matrix operand order repaired and checked by reading, not compiled, not run on hardware
//
#include <hip/hip_runtime.h>

// ---------------------------------------------------------------------------
// NNConv (edge-conditioned conv), fused as one bf16 WMMA GEMM:
//   m = Z @ W'   with  Z[e, k*16+i] = efeat[e,k]*feat[src[e],i]  (K=0..255)
//                      Z[e, 256+i]  = feat[src[e],i]             (bias term)
//                      Z[e, 272..287] = 0                        (pad to 9*32)
//   W'[k*16+i, o] = We[k, i*16+o];  W'[256+i, o] = be[i*16+o];  rest 0
// then segment-sum m into out via run-length-merged f32 atomics (dst sorted),
// on top of out = feat + bias.
// A-fragments are built with packed bf16 math (v_pk_mul_bf16) and
// double-buffered to dodge the WMMA->VALU WAR hazard nops.
// ---------------------------------------------------------------------------

typedef __bf16 bf16_t;
typedef __attribute__((ext_vector_type(16))) __bf16 v16bf;
typedef __attribute__((ext_vector_type(8)))  float  v8f;

#define NFRAG 9                 // 9 x K=32 bf16 WMMA fragments = K 288
#define WP_ELEMS (NFRAG * 512)  // pre-swizzled B fragments: [frag][lane][16]

union V16 {                     // v16bf viewed as 8 packed-bf16 dwords
  v16bf v;
  unsigned int u[8];
};

// ---- prep: build pre-swizzled bf16 B fragments of W' into d_ws -------------
__global__ void build_wp_kernel(const float* __restrict__ We,
                                const float* __restrict__ be,
                                bf16_t* __restrict__ Wp) {
  int idx = blockIdx.x * blockDim.x + threadIdx.x;
  if (idx >= WP_ELEMS) return;
  // B 32x16 bf16 layout: lane = hi*16+col holds column `col`,
  // K = frag*32 + hi*8 + j (j<8), frag*32 + 16 + hi*8 + (j-8) (j>=8) (element j of the v16bf).
  int w    = idx >> 9;       // fragment
  int r    = idx & 511;
  int lane = r >> 4;
  int j    = r & 15;
  int col  = lane & 15;
  int hi   = lane >> 4;
  int k    = w * 32 + (j < 8 ? hi * 8 + j : 16 + hi * 8 + (j - 8));
  float v = 0.0f;
  if (k < 256)      v = We[(k >> 4) * 256 + (k & 15) * 16 + col]; // We[k_e, i*16+o]
  else if (k < 272) v = be[(k - 256) * 16 + col];
  Wp[idx] = (bf16_t)v;
}

// ---- out = feat + bias ------------------------------------------------------
__global__ void init_out_kernel(const float* __restrict__ feat,
                                const float* __restrict__ bias,
                                float* __restrict__ out, int total) {
  int idx = blockIdx.x * blockDim.x + threadIdx.x;
  if (idx < total) out[idx] = feat[idx] + bias[idx & 15];
}

// ---- main edge kernel: one 16-edge tile per wave iteration ------------------
__global__ void __launch_bounds__(256)
nnconv_edge_kernel(const float* __restrict__ feat,
                   const float* __restrict__ efeat,
                   const bf16_t* __restrict__ Wp,
                   const float* __restrict__ We,
                   const float* __restrict__ be,
                   const int* __restrict__ src,
                   const int* __restrict__ dst,
                   float* __restrict__ out,
                   int E, int numTiles, int fullTiles) {
  const int lane = threadIdx.x & 31;
  const int wave = blockIdx.x * (blockDim.x >> 5) + (threadIdx.x >> 5);
  const int numWaves = gridDim.x * (blockDim.x >> 5);
  const int m  = lane & 15;   // A-matrix row owned by this lane (pairwise)
  const int hi = lane >> 4;   // which K/row half this lane covers
  const int o  = lane & 15;   // C-matrix column owned by this lane

  // Constant B fragments: held in VGPRs for the whole kernel (2x b128 each).
  v16bf B[NFRAG];
  {
    const v16bf* bp = (const v16bf*)Wp;
#pragma unroll
    for (int w = 0; w < NFRAG; ++w) B[w] = bp[w * 32 + lane];
  }

  for (int t = wave; t < numTiles; t += numWaves) {
    const int base = t * 16;
    if (t < fullTiles) {
      // ---- load per-edge operands (A row for lanes m / m+16 = edge base+m)
      const int e = base + m;
      const int s = src[e];
      float ef[16];
      {
        const float4* p = (const float4*)(efeat + (size_t)e * 16);
        float4 a0 = p[0], a1 = p[1], a2 = p[2], a3 = p[3];
        ef[0]=a0.x;  ef[1]=a0.y;  ef[2]=a0.z;  ef[3]=a0.w;
        ef[4]=a1.x;  ef[5]=a1.y;  ef[6]=a1.z;  ef[7]=a1.w;
        ef[8]=a2.x;  ef[9]=a2.y;  ef[10]=a2.z; ef[11]=a2.w;
        ef[12]=a3.x; ef[13]=a3.y; ef[14]=a3.z; ef[15]=a3.w;
      }
      float h[8];   // this lane only needs h[8*hi .. 8*hi+7]
      {
        const float4* p = (const float4*)(feat + (size_t)s * 16 + hi * 8);
        float4 a0 = p[0], a1 = p[1];
        h[0]=a0.x; h[1]=a0.y; h[2]=a0.z; h[3]=a0.w;
        h[4]=a1.x; h[5]=a1.y; h[6]=a1.z; h[7]=a1.w;
      }
      // prefetch this wave's next tile of the streaming input
      const int nb = base + numWaves * 16;
      if (nb + 16 <= E) __builtin_prefetch(efeat + (size_t)(nb + m) * 16, 0, 1);

      // ---- one-time packed-bf16 conversions for this tile ----
      // efs[a] = splat pair (bf16(ef[a]), bf16(ef[a]))
      // hbf[p] = pair (bf16(h[2p]), bf16(h[2p+1]))
      unsigned int efs[16], hbf[4];
#pragma unroll
      for (int q = 0; q < 16; ++q)
        asm("v_cvt_pk_bf16_f32 %0, %1, %2"
            : "=v"(efs[q]) : "v"(ef[q]), "v"(ef[q]));
#pragma unroll
      for (int p = 0; p < 4; ++p)
        asm("v_cvt_pk_bf16_f32 %0, %1, %2"
            : "=v"(hbf[p]) : "v"(h[2 * p]), "v"(h[2 * p + 1]));

      // A fragment builder: elems 0..7 = ef[2w]*h[j], elems 8..15 = ef[2w+1]*h[j]
      V16 A[2];
      auto buildFrag = [&](V16& F, int w) {
#pragma unroll
        for (int p = 0; p < 4; ++p) {
          asm("v_pk_mul_bf16 %0, %1, %2"
              : "=v"(F.u[p])     : "v"(hbf[p]), "v"(efs[2 * w]));
          asm("v_pk_mul_bf16 %0, %1, %2"
              : "=v"(F.u[4 + p]) : "v"(hbf[p]), "v"(efs[2 * w + 1]));
        }
      };

      // ---- Z @ W' : 8 outer-product fragments + 1 bias fragment,
      // double-buffered so no WAR hazard nops between WMMA and A-builds.
      buildFrag(A[0], 0);
      v8f c = {};
#pragma unroll
      for (int w = 0; w < NFRAG; ++w) {
        c = __builtin_amdgcn_wmma_f32_16x16x32_bf16(false, A[w & 1].v, false,
                                                    B[w], (short)0, c,
                                                    false, false);
        if (w < 7) {
          buildFrag(A[(w + 1) & 1], w + 1);
        } else if (w == 7) {
          // bias fragment (goes into A[0], read at w==8):
          // K=256..271 carries h, K=272..287 zero pad
#pragma unroll
          for (int p = 0; p < 4; ++p) {
            A[0].u[p]     = hbf[p];
            A[0].u[4 + p] = 0u;
          }
        }
      }

      // ---- scatter-add: lane owns rows base+8*hi+v (v=0..7), column o.
      // dst is sorted -> merge equal-destination runs before issuing atomics.
      int dv[8];
      {
        const int4* p = (const int4*)(dst + base + hi * 8);
        int4 d0 = p[0], d1 = p[1];
        dv[0]=d0.x; dv[1]=d0.y; dv[2]=d0.z; dv[3]=d0.w;
        dv[4]=d1.x; dv[5]=d1.y; dv[6]=d1.z; dv[7]=d1.w;
      }
      float acc = c[0];
      int cur = dv[0];
#pragma unroll
      for (int v = 1; v < 8; ++v) {
        if (dv[v] == cur) {
          acc += c[v];
        } else {
          atomicAdd(out + (size_t)cur * 16 + o, acc);
          cur = dv[v];
          acc = c[v];
        }
      }
      atomicAdd(out + (size_t)cur * 16 + o, acc);
    } else {
      // ---- partial remainder tile (< 16 edges): scalar fallback, no WMMA
      const int rem = E - base;
      if (lane < rem) {
        const int e2 = base + lane;
        const int s2 = src[e2];
        const int d2 = dst[e2];
        float ef2[16], h2[16];
        for (int q = 0; q < 16; ++q) {
          ef2[q] = efeat[(size_t)e2 * 16 + q];
          h2[q]  = feat[(size_t)s2 * 16 + q];
        }
        for (int oo = 0; oo < 16; ++oo) {
          float accv = 0.0f;
          for (int i = 0; i < 16; ++i) {
            float wio = be[i * 16 + oo];
            for (int k = 0; k < 16; ++k) wio += ef2[k] * We[k * 256 + i * 16 + oo];
            accv += h2[i] * wio;
          }
          atomicAdd(out + (size_t)d2 * 16 + oo, accv);
        }
      }
    }
  }
}

extern "C" void kernel_launch(void* const* d_in, const int* in_sizes, int n_in,
                              void* d_out, int out_size, void* d_ws, size_t ws_size,
                              hipStream_t stream) {
  const float* feat  = (const float*)d_in[0];
  const float* efeat = (const float*)d_in[1];
  const float* We    = (const float*)d_in[2];
  const float* be    = (const float*)d_in[3];
  const float* bias  = (const float*)d_in[4];
  const int*   src   = (const int*)d_in[5];
  const int*   dst   = (const int*)d_in[6];
  float*  out = (float*)d_out;
  bf16_t* Wp  = (bf16_t*)d_ws;   // needs WP_ELEMS*2 = 9216 bytes of scratch

  const int N = in_sizes[0] / 16;
  const int E = in_sizes[5];

  // 1) pre-swizzle W' (We + be) into bf16 B fragments
  build_wp_kernel<<<(WP_ELEMS + 255) / 256, 256, 0, stream>>>(We, be, Wp);
  // 2) out = feat + bias (residual + bias baseline for the segment sum)
  init_out_kernel<<<(N * 16 + 255) / 256, 256, 0, stream>>>(feat, bias, out, N * 16);
  // 3) fused edge GEMM + aggregation
  const int numTiles  = (E + 15) / 16;
  const int fullTiles = E / 16;
  const int wavesPerBlock = 8;        // 256 threads, wave32
  const int tilesPerWave  = 4;        // amortize B-fragment load
  int blocks = (numTiles + wavesPerBlock * tilesPerWave - 1) /
               (wavesPerBlock * tilesPerWave);
  if (blocks < 1) blocks = 1;
  nnconv_edge_kernel<<<blocks, 256, 0, stream>>>(feat, efeat, Wp, We, be,
                                                 src, dst, out,
                                                 E, numTiles, fullTiles);
}
